// BiHgru2_1d_6949257085053
// MI455X (gfx1250) — matrix operand order repaired and checked by reading, not compiled, not run on hardware
//
#include <hip/hip_runtime.h>
#include <hip/hip_bf16.h>

// ---------------------------------------------------------------------------
// BiHgru2 forward for MI455X (gfx1250, wave32, WMMA + TDM)
//   N=2048 seq, B=4 batch, D=2048, E=2, H=1024
// GEMMs: split-bf16 (hi+lo) V_WMMA_F32_16X16X32_BF16, f32 accumulation,
// block tiles staged in LDS by the Tensor Data Mover (tensor_load_to_lds),
// double-buffered with s_wait_tensorcnt.
// ---------------------------------------------------------------------------

#define SEQ_N 2048
#define BATCH 4
#define DIM   2048
#define HHEAD 1024
#define MROWS (SEQ_N * BATCH)   // 8192
#define NC1   (3 * DIM)         // 6144
#define KDIM  DIM               // 2048

#define BM 64
#define BN 128
#define BK 32

typedef __attribute__((ext_vector_type(16))) __bf16 bf16x16;
typedef __attribute__((ext_vector_type(8)))  __bf16 bf16x8;
typedef __attribute__((ext_vector_type(8)))  float  f32x8;
typedef unsigned int u32x4 __attribute__((ext_vector_type(4)));
typedef int          i32x8 __attribute__((ext_vector_type(8)));
typedef int          i32x4 __attribute__((ext_vector_type(4)));

// ------------------------- fp32 -> bf16 hi/lo split -------------------------
__global__ void split_f32(const float* __restrict__ src,
                          __bf16* __restrict__ hi, __bf16* __restrict__ lo,
                          long n) {
  long i = (long)blockIdx.x * blockDim.x + threadIdx.x;
  if (i >= n) return;
  float v = src[i];
  __bf16 h = (__bf16)v;
  hi[i] = h;
  lo[i] = (__bf16)(v - (float)h);
}

// --------------------- Tensor Data Mover 2D tile load -----------------------
// Loads a (tile_rows x tile_cols) bf16 tile (row stride = stride0 elements)
// from global memory into LDS at byte offset lds_byte, packed row-major.
// D# per CDNA5 ISA ch.8: group0 = {count,lds,global,type}, group1 = dims.
static __device__ inline void tdm_load_tile_bf16(unsigned lds_byte,
                                                 const __bf16* gptr,
                                                 unsigned tile_cols,
                                                 unsigned tile_rows,
                                                 unsigned tensor_d0,
                                                 unsigned tensor_d1,
                                                 unsigned long long stride0) {
  unsigned long long ga = (unsigned long long)(uintptr_t)gptr;
  u32x4 g0;
  g0[0] = 1u;                                   // count=1 (valid), user mode
  g0[1] = lds_byte;                             // lds_addr
  g0[2] = (unsigned)ga;                         // global_addr[31:0]
  g0[3] = (unsigned)((ga >> 32) & 0x01FFFFFFu)  // global_addr[56:32]
          | (2u << 30);                         // type=2 ("image")
  i32x8 g1;
  g1[0] = (int)(1u << 16);                      // wg_mask=0, data_size=1 (2B)
  g1[1] = (int)((tensor_d0 & 0xFFFFu) << 16);   // tensor_dim0[15:0]
  g1[2] = (int)((tensor_d0 >> 16) | ((tensor_d1 & 0xFFFFu) << 16));
  g1[3] = (int)((tensor_d1 >> 16) | (tile_cols << 16));   // tile_dim0
  g1[4] = (int)tile_rows;                       // tile_dim1 (tile_dim2=0)
  g1[5] = (int)(unsigned)(stride0 & 0xFFFFFFFFull);       // dim0 stride lo
  g1[6] = (int)(unsigned)((stride0 >> 32) & 0xFFFFull);   // dim0 stride hi
  g1[7] = 0;
  i32x4 z4 = {0, 0, 0, 0};
  i32x8 z8 = {0, 0, 0, 0, 0, 0, 0, 0};
  __builtin_amdgcn_tensor_load_to_lds(g0, g1, z4, z4, z8, 0);
}

static __device__ inline unsigned lds_off(const void* p) {
  // flat LDS address: aperture in [63:32], workgroup-relative offset in [31:0]
  return (unsigned)(uintptr_t)p;
}

// ------------------------- WMMA fragment helpers ----------------------------
// A (16x32 bf16): lane m = l&15, half = l>>4;
//   elems [0..15] = K(k0+half*16..)
static __device__ inline bf16x16 load_a_frag_lds(const __bf16* rowPtr, int half) {
  bf16x8 a = *reinterpret_cast<const bf16x8*>(rowPtr + half * 16);
  bf16x8 b = *reinterpret_cast<const bf16x8*>(rowPtr + half * 16 + 8);
  bf16x16 r;
#pragma unroll
  for (int i = 0; i < 8; ++i) { r[i] = a[i]; r[i + 8] = b[i]; }
  return r;
}

// --------------------- split-bf16 GEMM: C = A @ W^T + bias ------------------
// A: (M x K) bf16 hi/lo, W: (Nc x K) bf16 hi/lo, C: (M x Nc) f32.
// Block (256 thr, 8 waves) computes BM x BN; wave computes 32x32 (2x2 WMMA).
// Block tiles of A (BMxBK) and W (BNxBK), hi+lo, DMA'd to LDS by the TDM,
// double buffered.
__global__ __launch_bounds__(256)
void gemm_bf16split_tdm(const __bf16* __restrict__ Ah, const __bf16* __restrict__ Al,
                        const __bf16* __restrict__ Wh, const __bf16* __restrict__ Wl,
                        const float* __restrict__ bias, float* __restrict__ C,
                        int M, int Nc, int K) {
  __shared__ __bf16 sAh[2][BM * BK];   //  8 KB
  __shared__ __bf16 sAl[2][BM * BK];   //  8 KB
  __shared__ __bf16 sBh[2][BN * BK];   // 16 KB
  __shared__ __bf16 sBl[2][BN * BK];   // 16 KB

  const int tilesN = Nc / BN;
  const int bm0 = (blockIdx.x / tilesN) * BM;
  const int bn0 = (blockIdx.x % tilesN) * BN;

  const int waveId = __builtin_amdgcn_readfirstlane((int)(threadIdx.x >> 5));
  const int lane = threadIdx.x & 31;
  const int m15 = lane & 15;
  const int half = lane >> 4;
  const int wrow = (waveId >> 2) * 32;   // 0 / 32   within BM
  const int wcol = (waveId & 3) * 32;    // 0..96    within BN

  const int KSTEPS = K / BK;

  // ---- prologue: stage 0 DMA (wave 0 only; TDM ignores EXEC, so the issue
  // path must be a uniform scalar branch) ----
  if (waveId == 0) {
    const __bf16* a0h = Ah + (size_t)bm0 * K;
    const __bf16* a0l = Al + (size_t)bm0 * K;
    const __bf16* b0h = Wh + (size_t)bn0 * K;
    const __bf16* b0l = Wl + (size_t)bn0 * K;
    tdm_load_tile_bf16(lds_off(&sAh[0][0]), a0h, BK, BM, K, M,  K);
    tdm_load_tile_bf16(lds_off(&sAl[0][0]), a0l, BK, BM, K, M,  K);
    tdm_load_tile_bf16(lds_off(&sBh[0][0]), b0h, BK, BN, K, Nc, K);
    tdm_load_tile_bf16(lds_off(&sBl[0][0]), b0l, BK, BN, K, Nc, K);
  }

  f32x8 acc[2][2] = {};

  for (int i = 0; i < KSTEPS; ++i) {
    if (waveId == 0) {
      if (i + 1 < KSTEPS) {
        const int kk = (i + 1) * BK;
        const int nb = (i + 1) & 1;
        tdm_load_tile_bf16(lds_off(&sAh[nb][0]), Ah + (size_t)bm0 * K + kk, BK, BM, K, M,  K);
        tdm_load_tile_bf16(lds_off(&sAl[nb][0]), Al + (size_t)bm0 * K + kk, BK, BM, K, M,  K);
        tdm_load_tile_bf16(lds_off(&sBh[nb][0]), Wh + (size_t)bn0 * K + kk, BK, BN, K, Nc, K);
        tdm_load_tile_bf16(lds_off(&sBl[nb][0]), Wl + (size_t)bn0 * K + kk, BK, BN, K, Nc, K);
        __builtin_amdgcn_s_wait_tensorcnt(4);   // stage i complete (in-order)
      } else {
        __builtin_amdgcn_s_wait_tensorcnt(0);
      }
    }
    __syncthreads();                            // stage i visible to all waves

    const int cur = i & 1;
    bf16x16 ah[2], al[2], bh[2], bl[2];
#pragma unroll
    for (int mt = 0; mt < 2; ++mt) {
      const __bf16* rh = &sAh[cur][(wrow + mt * 16 + m15) * BK];
      const __bf16* rl = &sAl[cur][(wrow + mt * 16 + m15) * BK];
      ah[mt] = load_a_frag_lds(rh, half);
      al[mt] = load_a_frag_lds(rl, half);
    }
#pragma unroll
    for (int nt = 0; nt < 2; ++nt) {
      const int rb = (wcol + nt * 16 + m15) * BK + 16 * half;
      bh[nt] = *reinterpret_cast<const bf16x16*>(&sBh[cur][rb]);
      bl[nt] = *reinterpret_cast<const bf16x16*>(&sBl[cur][rb]);
    }
#pragma unroll
    for (int mt = 0; mt < 2; ++mt) {
#pragma unroll
      for (int nt = 0; nt < 2; ++nt) {
        acc[mt][nt] = __builtin_amdgcn_wmma_f32_16x16x32_bf16(
            false, ah[mt], false, bh[nt], (short)0, acc[mt][nt], false, false);
        acc[mt][nt] = __builtin_amdgcn_wmma_f32_16x16x32_bf16(
            false, ah[mt], false, bl[nt], (short)0, acc[mt][nt], false, false);
        acc[mt][nt] = __builtin_amdgcn_wmma_f32_16x16x32_bf16(
            false, al[mt], false, bh[nt], (short)0, acc[mt][nt], false, false);
      }
    }
    __syncthreads();                            // all reads of buf done
  }

  // C layout: VGPR r, lanes 0-15 -> M=r, lanes 16-31 -> M=r+8; N = lane&15
#pragma unroll
  for (int mt = 0; mt < 2; ++mt) {
#pragma unroll
    for (int nt = 0; nt < 2; ++nt) {
      const int col = bn0 + wcol + nt * 16 + m15;
      const float bv = bias ? bias[col] : 0.0f;
#pragma unroll
      for (int r = 0; r < 8; ++r) {
        const int row = bm0 + wrow + mt * 16 + half * 8 + r;
        C[(size_t)row * (size_t)Nc + col] = acc[mt][nt][r] + bv;
      }
    }
  }
}

// ---------------- bidirectional linear scan + og contraction ----------------
static __device__ inline float sigm(float x) { return 1.0f / (1.0f + __expf(-x)); }

__global__ void scan_kernel(const float* __restrict__ feat,
                            float* __restrict__ o_part) {
  const int tid = blockIdx.x * blockDim.x + threadIdx.x;
  if (tid >= BATCH * HHEAD) return;
  const int h = tid & (HHEAD - 1);
  const int b = tid >> 10;
  const size_t cI = 2 * h;
  const size_t cO = DIM + 2 * h;
  const size_t cF = 2 * DIM + 2 * h;

  float s00 = 0.f, s01 = 0.f, s10 = 0.f, s11 = 0.f;
  for (int t = 0; t < SEQ_N; ++t) {
    const size_t row = (size_t)(t * BATCH + b) * NC1;
    float2 iv = *reinterpret_cast<const float2*>(feat + row + cI);
    float2 gv = *reinterpret_cast<const float2*>(feat + row + cO);
    float2 fv = *reinterpret_cast<const float2*>(feat + row + cF);
    const float a0 = iv.x * sigm(iv.x), a1 = iv.y * sigm(iv.y);   // silu
    const float og0 = sigm(gv.x), og1 = sigm(gv.y);
    const float l0 = sigm(fv.x), l1 = sigm(fv.y);
    s00 = l0 * s00 + (1.f - l0) * a0;
    s01 = l0 * s01 + (1.f - l0) * a1;
    s10 = l1 * s10 + (1.f - l1) * a0;
    s11 = l1 * s11 + (1.f - l1) * a1;
    float2 o;
    o.x = s00 * og0 + s10 * og1;
    o.y = s01 * og0 + s11 * og1;
    *reinterpret_cast<float2*>(o_part + (size_t)(t * BATCH + b) * DIM + 2 * h) = o;
  }

  s00 = s01 = s10 = s11 = 0.f;
  for (int t = SEQ_N - 1; t >= 0; --t) {
    const size_t row = (size_t)(t * BATCH + b) * NC1;
    float2 iv = *reinterpret_cast<const float2*>(feat + row + cI);
    float2 gv = *reinterpret_cast<const float2*>(feat + row + cO);
    float2 fv = *reinterpret_cast<const float2*>(feat + row + cF);
    const float a0 = iv.x * sigm(iv.x), a1 = iv.y * sigm(iv.y);
    const float og0 = sigm(gv.x), og1 = sigm(gv.y);
    const float l0 = sigm(fv.x), l1 = sigm(fv.y);
    s00 = l0 * s00 + (1.f - l0) * a0;
    s01 = l0 * s01 + (1.f - l0) * a1;
    s10 = l1 * s10 + (1.f - l1) * a0;
    s11 = l1 * s11 + (1.f - l1) * a1;
    float2* op = reinterpret_cast<float2*>(o_part + (size_t)(t * BATCH + b) * DIM + 2 * h);
    float2 v = *op;
    v.x += s00 * og0 + s10 * og1;
    v.y += s01 * og0 + s11 * og1;
    *op = v;
  }
}

// ------------------- LayerNorm fused with hi/lo bf16 split ------------------
__global__ __launch_bounds__(256)
void ln_split(const float* __restrict__ o, const float* __restrict__ gamma,
              const float* __restrict__ beta,
              __bf16* __restrict__ hi, __bf16* __restrict__ lo) {
  __shared__ float red[256];
  const int row = blockIdx.x;
  const int t = threadIdx.x;
  const float* p = o + (size_t)row * DIM;

  float vals[8];
  float s = 0.f;
#pragma unroll
  for (int i = 0; i < 8; ++i) { vals[i] = p[t + i * 256]; s += vals[i]; }
  red[t] = s; __syncthreads();
  for (int off = 128; off > 0; off >>= 1) {
    if (t < off) red[t] += red[t + off];
    __syncthreads();
  }
  const float mu = red[0] * (1.0f / DIM);
  __syncthreads();

  float q = 0.f;
#pragma unroll
  for (int i = 0; i < 8; ++i) { float d = vals[i] - mu; q += d * d; }
  red[t] = q; __syncthreads();
  for (int off = 128; off > 0; off >>= 1) {
    if (t < off) red[t] += red[t + off];
    __syncthreads();
  }
  const float rs = rsqrtf(red[0] * (1.0f / DIM) + 1e-5f);

#pragma unroll
  for (int i = 0; i < 8; ++i) {
    const int c = t + i * 256;
    const float y = (vals[i] - mu) * rs * gamma[c] + beta[c];
    const __bf16 h = (__bf16)y;
    hi[(size_t)row * DIM + c] = h;
    lo[(size_t)row * DIM + c] = (__bf16)(y - (float)h);
  }
}

// ---------------------------------------------------------------------------
extern "C" void kernel_launch(void* const* d_in, const int* in_sizes, int n_in,
                              void* d_out, int out_size, void* d_ws, size_t ws_size,
                              hipStream_t stream) {
  const float* x     = (const float*)d_in[0];   // (N,B,D)
  const float* W_in  = (const float*)d_in[1];   // (3D,D)
  const float* b_in  = (const float*)d_in[2];   // (3D)
  const float* gamma = (const float*)d_in[3];   // (D)
  const float* beta  = (const float*)d_in[4];   // (D)
  const float* W_out = (const float*)d_in[5];   // (D,D)
  const float* b_out = (const float*)d_in[6];   // (D)
  float* out = (float*)d_out;                   // (N,B,D) f32

  // ---- workspace layout ----
  char* ws = (char*)d_ws;
  const size_t nX  = (size_t)MROWS * KDIM;
  const size_t nWi = (size_t)NC1 * KDIM;
  const size_t nWo = (size_t)DIM * KDIM;
  size_t off = 0;
  __bf16* x_hi   = (__bf16*)(ws + off); off += nX  * 2;
  __bf16* x_lo   = (__bf16*)(ws + off); off += nX  * 2;
  __bf16* wi_hi  = (__bf16*)(ws + off); off += nWi * 2;
  __bf16* wi_lo  = (__bf16*)(ws + off); off += nWi * 2;
  __bf16* wo_hi  = (__bf16*)(ws + off); off += nWo * 2;
  __bf16* wo_lo  = (__bf16*)(ws + off); off += nWo * 2;
  float*  feat   = (float*)(ws + off);  off += (size_t)MROWS * NC1 * 4;
  float*  o_part = (float*)(ws + off);  off += (size_t)MROWS * DIM * 4;
  __bf16* o_hi   = (__bf16*)(ws + off); off += nX * 2;
  __bf16* o_lo   = (__bf16*)(ws + off); off += nX * 2;

  // 1) fp32 -> bf16 hi/lo splits
  split_f32<<<(int)((nX  + 255) / 256), 256, 0, stream>>>(x,     x_hi,  x_lo,  (long)nX);
  split_f32<<<(int)((nWi + 255) / 256), 256, 0, stream>>>(W_in,  wi_hi, wi_lo, (long)nWi);
  split_f32<<<(int)((nWo + 255) / 256), 256, 0, stream>>>(W_out, wo_hi, wo_lo, (long)nWo);

  // 2) feat = x @ W_in^T + b_in   (8192 x 6144, K=2048)
  gemm_bf16split_tdm<<<(MROWS / BM) * (NC1 / BN), 256, 0, stream>>>(
      x_hi, x_lo, wi_hi, wi_lo, b_in, feat, MROWS, NC1, KDIM);

  // 3) bidirectional scan + gate contraction -> o_part (8192 x 2048)
  scan_kernel<<<(BATCH * HHEAD) / 256, 256, 0, stream>>>(feat, o_part);

  // 4) LayerNorm + hi/lo split
  ln_split<<<MROWS, 256, 0, stream>>>(o_part, gamma, beta, o_hi, o_lo);

  // 5) out = o_ln @ W_out^T + b_out  (8192 x 2048, K=2048)
  gemm_bf16split_tdm<<<(MROWS / BM) * (DIM / BN), 256, 0, stream>>>(
      o_hi, o_lo, wo_hi, wo_lo, b_out, out, MROWS, DIM, KDIM);
}
